// GIN_43894565765481
// MI455X (gfx1250) — matrix operand order repaired and checked by reading, not compiled, not run on hardware
//
#include <hip/hip_runtime.h>

#define N_NODESC 100000
#define N_EDGESC 1600000
#define NG 128
#define HD 64
#define GDIM 192
#define LRELU 0.01f
#define GRID_BLK 640

typedef __attribute__((ext_vector_type(16))) _Float16 v16h;
typedef __attribute__((ext_vector_type(8)))  float    v8f;

union Frag16 { v16h v; uint4 q[2]; };

// leaky_relu(x) == max(x, 0.01*x) exactly (slope < 1): v_mul + v_max, no cndmask
__device__ __forceinline__ float lrelu(float x) { return fmaxf(x, LRELU * x); }

// ---------------- ordered-int float max mapping (for atomic max) -------------
__device__ __forceinline__ unsigned ordkey(float f) {
  unsigned u = __float_as_uint(f);
  return (u & 0x80000000u) ? ~u : (u | 0x80000000u);
}
__device__ __forceinline__ float ordun(unsigned k) {
  unsigned u = (k & 0x80000000u) ? (k ^ 0x80000000u) : ~k;
  return __uint_as_float(u);
}
#define ORD_NEG_INF 0x007FFFFFu  // ordkey(-inf)

// ---------------- WMMA fragment helpers (CDNA5 16x16x32 f16) -----------------
// B-fragment swizzled storage: weight element (k,n) of a KxN (row-major [K][64])
// matrix lives at frag=(k/32)*4+(n/16), lane=(n&15)+16*((k%32)/16), j=k%16.
// Per (frag,lane) the 16 halfs are CONTIGUOUS -> two ds_load_b128 (LICM-hoisted
// into registers across the tile loop).
__device__ __forceinline__ int bfrag_pos(int k, int n) {
  const int kt = k >> 5, kk = k & 31, hi = (kk >> 4) & 1;
  const int lane = (n & 15) + (hi << 4);
  const int nt = n >> 4;
  const int j = kk & 15;
  return ((((kt << 2) + nt) << 5) + lane) * 16 + j;
}
__device__ __forceinline__ v16h load_bfrag(const _Float16* sWf, int frag, int lane) {
  const _Float16* p = sWf + ((frag << 5) + lane) * 16;
  Frag16 f;
  f.q[0] = *(const uint4*)p;
  f.q[1] = *(const uint4*)(p + 8);
  return f.v;
}
// A-fragment from a row-major 16xHD f16 tile (LDS or global): lane m=l&15 holds
// row m; kh=16*(l>>4); a[0..15]=row[kt*32+kh .. kt*32+kh+15].
// Both halves are contiguous 16B -> b128 loads.
__device__ __forceinline__ v16h frag_A_mem(const _Float16* row0p, int lane, int kt) {
  const int m = lane & 15, kh = (lane >> 4) << 4;
  const _Float16* p = row0p + m * HD + kt * 32 + kh;
  Frag16 f;
  f.q[0] = *(const uint4*)p;
  f.q[1] = *(const uint4*)(p + 8);
  return f.v;
}
__device__ __forceinline__ v8f wmma32(v16h a, v16h b, v8f c) {
  return __builtin_amdgcn_wmma_f32_16x16x32_f16(false, a, false, b, (short)0, c,
                                                false, false);
}
// C/D: vgpr r, lanes 0-15 -> (M=r, N=lane); lanes 16-31 -> (M=r+8, N=lane-16)
template <int ACT>  // 0 none, 1 relu, 2 leaky
__device__ __forceinline__ void stash_C(_Float16* act, int lane, int nt, v8f c, float b) {
  const int n = lane & 15, moff = (lane >> 4) << 3;
#pragma unroll
  for (int r = 0; r < 8; ++r) {
    float v = c[r] + b;
    if (ACT == 1) v = fmaxf(v, 0.f);
    if (ACT == 2) v = lrelu(v);
    act[(moff + r) * HD + nt * 16 + n] = (_Float16)v;
  }
}
template <int ACT>
__device__ __forceinline__ void wave_gemm64(const _Float16* aIn, const _Float16* sWf,
                                            const float* bias4, _Float16* aOut, int lane) {
  v16h a0 = frag_A_mem(aIn, lane, 0);
  v16h a1 = frag_A_mem(aIn, lane, 1);
#pragma unroll
  for (int nt = 0; nt < 4; ++nt) {
    v8f c = {};
    c = wmma32(a0, load_bfrag(sWf, nt, lane), c);
    c = wmma32(a1, load_bfrag(sWf, 4 + nt, lane), c);
    stash_C<ACT>(aOut, lane, nt, c, bias4[nt]);
  }
}

// ---------------- 3-layer embedding MLP (nodes: INC=4, edges: INC=8) ---------
template <int INC, bool OUT_F16>
__global__ __launch_bounds__(256) void embed_mlp_kernel(
    const float* __restrict__ X, const float* __restrict__ W1, const float* __restrict__ b1,
    const float* __restrict__ W2, const float* __restrict__ b2,
    const float* __restrict__ W3, const float* __restrict__ b3,
    float* __restrict__ outF32, _Float16* __restrict__ outF16, int nrows) {
  __shared__ _Float16 sW1f[4 * 32 * 16];                 // K padded to 32, swizzled
  __shared__ _Float16 sW2f[8 * 32 * 16], sW3f[8 * 32 * 16];
  __shared__ float    sB[3 * HD];
  __shared__ _Float16 sAct[8][2][16 * HD];               // per-wave ping/pong
  const int tid = threadIdx.x;
  for (int i = tid; i < 32 * HD; i += 256) {
    int k = i / HD, n = i % HD;
    sW1f[bfrag_pos(k, n)] = (_Float16)(k < INC ? W1[k * HD + n] : 0.f);
  }
  for (int i = tid; i < HD * HD; i += 256) {
    int k = i / HD, n = i % HD, p = bfrag_pos(k, n);
    sW2f[p] = (_Float16)W2[i];
    sW3f[p] = (_Float16)W3[i];
  }
  for (int i = tid; i < HD; i += 256) { sB[i] = b1[i]; sB[HD + i] = b2[i]; sB[2 * HD + i] = b3[i]; }
  __syncthreads();

  const int wave = tid >> 5, lane = tid & 31;
  const int n16 = lane & 15, moff = (lane >> 4) << 3;
  float b1v[4], b2v[4], b3v[4];
#pragma unroll
  for (int nt = 0; nt < 4; ++nt) {
    b1v[nt] = sB[nt * 16 + n16];
    b2v[nt] = sB[HD + nt * 16 + n16];
    b3v[nt] = sB[2 * HD + nt * 16 + n16];
  }
  _Float16* aA = sAct[wave][0];
  _Float16* aB = sAct[wave][1];
  const int ntiles = nrows >> 4;
  for (int tile = blockIdx.x * 8 + wave; tile < ntiles; tile += gridDim.x * 8) {
    const int row0 = tile << 4;
    const int tnext = tile + gridDim.x * 8;
    if (tnext < ntiles) __builtin_prefetch(X + (size_t)(tnext << 4) * INC, 0, 1);
    // ---- layer 1: A fragment straight from global (zero-padded to K=32) ----
    v16h a0 = {};
    if (lane < 16) {
      const float4* xr = (const float4*)(X + (size_t)row0 * INC + lane * INC);
      float4 p0 = xr[0];
      a0[0] = (_Float16)p0.x; a0[1] = (_Float16)p0.y;
      a0[2] = (_Float16)p0.z; a0[3] = (_Float16)p0.w;
      if (INC == 8) {
        float4 p1 = xr[1];
        a0[4] = (_Float16)p1.x; a0[5] = (_Float16)p1.y;
        a0[6] = (_Float16)p1.z; a0[7] = (_Float16)p1.w;
      }
    }
#pragma unroll
    for (int nt = 0; nt < 4; ++nt) {
      v8f c = {};
      c = wmma32(a0, load_bfrag(sW1f, nt, lane), c);
      stash_C<1>(aA, lane, nt, c, b1v[nt]);
    }
    // ---- layer 2 (+ReLU) ----
    wave_gemm64<1>(aA, sW2f, b2v, aB, lane);
    // ---- layer 3 (linear) straight to global ----
    v16h x0 = frag_A_mem(aB, lane, 0), x1 = frag_A_mem(aB, lane, 1);
#pragma unroll
    for (int nt = 0; nt < 4; ++nt) {
      v8f c = {};
      c = wmma32(x0, load_bfrag(sW3f, nt, lane), c);
      c = wmma32(x1, load_bfrag(sW3f, 4 + nt, lane), c);
      const float bb = b3v[nt];
#pragma unroll
      for (int r = 0; r < 8; ++r) {
        float v = c[r] + bb;
        size_t o = (size_t)(row0 + moff + r) * HD + nt * 16 + n16;
        if (OUT_F16) outF16[o] = (_Float16)v; else outF32[o] = v;
      }
    }
  }
}

// ---------------- GINE 2-layer MLP: h = leaky(z@W1+b1)@W2 + b2 ---------------
__global__ __launch_bounds__(256) void gine_mlp_kernel(
    const _Float16* __restrict__ Z, const float* __restrict__ W1, const float* __restrict__ b1,
    const float* __restrict__ W2, const float* __restrict__ b2,
    float* __restrict__ Hout, int nrows) {
  __shared__ _Float16 sW1f[8 * 32 * 16], sW2f[8 * 32 * 16];
  __shared__ float    sB[2 * HD];
  __shared__ _Float16 sAct[8][16 * HD];
  const int tid = threadIdx.x;
  for (int i = tid; i < HD * HD; i += 256) {
    int k = i / HD, n = i % HD, p = bfrag_pos(k, n);
    sW1f[p] = (_Float16)W1[i];
    sW2f[p] = (_Float16)W2[i];
  }
  for (int i = tid; i < HD; i += 256) { sB[i] = b1[i]; sB[HD + i] = b2[i]; }
  __syncthreads();
  const int wave = tid >> 5, lane = tid & 31;
  const int n16 = lane & 15, moff = (lane >> 4) << 3;
  float b1v[4], b2v[4];
#pragma unroll
  for (int nt = 0; nt < 4; ++nt) {
    b1v[nt] = sB[nt * 16 + n16];
    b2v[nt] = sB[HD + nt * 16 + n16];
  }
  _Float16* aA = sAct[wave];
  const int ntiles = nrows >> 4;
  for (int tile = blockIdx.x * 8 + wave; tile < ntiles; tile += gridDim.x * 8) {
    const int row0 = tile << 4;
    const int tnext = tile + gridDim.x * 8;
    if (tnext < ntiles) __builtin_prefetch(Z + (size_t)(tnext << 4) * HD, 0, 1);
    // layer 1: A fragments directly from global f16 (row-major, 16B aligned)
    const _Float16* zrow = Z + (size_t)row0 * HD;
    v16h a0 = frag_A_mem(zrow, lane, 0);
    v16h a1 = frag_A_mem(zrow, lane, 1);
#pragma unroll
    for (int nt = 0; nt < 4; ++nt) {
      v8f c = {};
      c = wmma32(a0, load_bfrag(sW1f, nt, lane), c);
      c = wmma32(a1, load_bfrag(sW1f, 4 + nt, lane), c);
      stash_C<2>(aA, lane, nt, c, b1v[nt]);  // leaky_relu
    }
    // layer 2 (linear) to global f32
    v16h x0 = frag_A_mem(aA, lane, 0), x1 = frag_A_mem(aA, lane, 1);
#pragma unroll
    for (int nt = 0; nt < 4; ++nt) {
      v8f c = {};
      c = wmma32(x0, load_bfrag(sW2f, nt, lane), c);
      c = wmma32(x1, load_bfrag(sW2f, 4 + nt, lane), c);
      const float bb = b2v[nt];
#pragma unroll
      for (int r = 0; r < 8; ++r)
        Hout[(size_t)(row0 + moff + r) * HD + nt * 16 + n16] = c[r] + bb;
    }
  }
}

// ---------------- CSR build (count / scan / fill) ----------------------------
__global__ void init_kernel(int* __restrict__ deg, unsigned* __restrict__ pk) {
  int i = blockIdx.x * 256 + threadIdx.x;
  if (i <= N_NODESC) deg[i] = 0;
  if (i < NG * GDIM) pk[i] = ORD_NEG_INF;
}
__global__ void count_kernel(const int* __restrict__ dst, int* __restrict__ deg) {
  int e = blockIdx.x * 256 + threadIdx.x;
  if (e < N_EDGESC) atomicAdd(&deg[dst[e]], 1);
}
__global__ __launch_bounds__(1024) void scan1_kernel(const int* __restrict__ deg,
                                                     int* __restrict__ excl,
                                                     int* __restrict__ parts) {
  __shared__ int s[1024];
  int i = blockIdx.x * 1024 + threadIdx.x;
  int v = (i < N_NODESC) ? deg[i] : 0;
  s[threadIdx.x] = v;
  __syncthreads();
  for (int ofs = 1; ofs < 1024; ofs <<= 1) {
    int t = (threadIdx.x >= ofs) ? s[threadIdx.x - ofs] : 0;
    __syncthreads();
    s[threadIdx.x] += t;
    __syncthreads();
  }
  if (i < N_NODESC) excl[i] = s[threadIdx.x] - v;
  if (threadIdx.x == 1023) parts[blockIdx.x] = s[1023];
}
__global__ __launch_bounds__(1024) void scan2_kernel(int* __restrict__ parts, int nblk) {
  __shared__ int s[1024];
  int v = (threadIdx.x < nblk) ? parts[threadIdx.x] : 0;
  s[threadIdx.x] = v;
  __syncthreads();
  for (int ofs = 1; ofs < 1024; ofs <<= 1) {
    int t = (threadIdx.x >= ofs) ? s[threadIdx.x - ofs] : 0;
    __syncthreads();
    s[threadIdx.x] += t;
    __syncthreads();
  }
  if (threadIdx.x < nblk) parts[threadIdx.x] = s[threadIdx.x] - v;
}
__global__ __launch_bounds__(1024) void scan3_kernel(const int* __restrict__ excl,
                                                     const int* __restrict__ parts,
                                                     int* __restrict__ rowptr) {
  int i = blockIdx.x * 1024 + threadIdx.x;
  if (i < N_NODESC) rowptr[i] = excl[i] + parts[blockIdx.x];
  if (i == 0) rowptr[N_NODESC] = N_EDGESC;
}
__global__ void cursor_kernel(const int* __restrict__ rowptr, int* __restrict__ cur) {
  int i = blockIdx.x * 256 + threadIdx.x;
  if (i < N_NODESC) cur[i] = rowptr[i];
}
__global__ void fill_kernel(const int* __restrict__ dst, int* __restrict__ cur,
                            int* __restrict__ eid) {
  int e = blockIdx.x * 256 + threadIdx.x;
  if (e < N_EDGESC) {
    int p = atomicAdd(&cur[dst[e]], 1);
    eid[p] = e;
  }
}

// ------------- per-layer aggregation: z = (1+eps)*h + segmax(leaky(h[s]+ea)) -
__global__ __launch_bounds__(256) void agg_kernel(
    const int* __restrict__ rowptr, const int* __restrict__ eid,
    const int* __restrict__ src, const _Float16* __restrict__ EA,
    const float* __restrict__ Hin, const float* __restrict__ epsArr, int l,
    _Float16* __restrict__ Z) {
  const int v = blockIdx.x * 8 + (threadIdx.x >> 5);
  if (v >= N_NODESC) return;
  const int lane = threadIdx.x & 31;
  const int beg = rowptr[v], end = rowptr[v + 1];
  float m0 = -__builtin_inff(), m1 = -__builtin_inff();
  for (int k = beg; k < end; ++k) {
    const int e = eid[k];
    const int s = src[e];
    const float2 hv = ((const float2*)(Hin + (size_t)s * HD))[lane];
    union { unsigned u; _Float16 h2[2]; } ev;
    ev.u = ((const unsigned*)(EA + (size_t)e * HD))[lane];
    m0 = fmaxf(m0, lrelu(hv.x + (float)ev.h2[0]));
    m1 = fmaxf(m1, lrelu(hv.y + (float)ev.h2[1]));
  }
  if (!(m0 >= -3.0e38f && m0 <= 3.0e38f)) m0 = 0.f;  // isolated / non-finite -> 0
  if (!(m1 >= -3.0e38f && m1 <= 3.0e38f)) m1 = 0.f;
  const float e1 = 1.f + epsArr[l];
  const float2 hv = ((const float2*)(Hin + (size_t)v * HD))[lane];
  union { unsigned u; _Float16 h2[2]; } oz;
  oz.h2[0] = (_Float16)(e1 * hv.x + m0);
  oz.h2[1] = (_Float16)(e1 * hv.y + m1);
  ((unsigned*)(Z + (size_t)v * HD))[lane] = oz.u;
}

// ---------------- graph pooling: segment_max(h, batch) -----------------------
__global__ void pool_kernel(const float* __restrict__ Hf, const int* __restrict__ batch,
                            unsigned* __restrict__ pk, int l) {
  const long idx = (long)blockIdx.x * 256 + threadIdx.x;
  const int v = (int)(idx >> 5);
  if (v >= N_NODESC) return;
  const int q = (int)(idx & 31);
  const int g = batch[v];
  const float2 hv = ((const float2*)(Hf + (size_t)v * HD))[q];
  atomicMax(&pk[(size_t)g * GDIM + l * HD + 2 * q],     ordkey(hv.x));
  atomicMax(&pk[(size_t)g * GDIM + l * HD + 2 * q + 1], ordkey(hv.y));
}

// ---------------- output MLP --------------------------------------------------
__global__ __launch_bounds__(GDIM) void out1_kernel(
    const unsigned* __restrict__ pk, const float* __restrict__ oW1,
    const float* __restrict__ ob1, const float* __restrict__ gamma,
    const float* __restrict__ beta, float* __restrict__ Y) {
  __shared__ float hp[GDIM];
  const int g = blockIdx.x, j = threadIdx.x;
  hp[j] = ordun(pk[g * GDIM + j]);
  __syncthreads();
  float acc = ob1[j];
  for (int k = 0; k < GDIM; ++k) acc += hp[k] * oW1[k * GDIM + j];
  acc = gamma[j] * acc + beta[j];
  acc = lrelu(acc);
  Y[g * GDIM + j] = acc;
}
__global__ __launch_bounds__(NG) void out2_kernel(const float* __restrict__ Y,
                                                  const float* __restrict__ oW2,
                                                  const float* __restrict__ ob2,
                                                  float* __restrict__ out) {
  const int g = threadIdx.x;
  float acc = ob2[0];
  for (int k = 0; k < GDIM; ++k) acc += Y[g * GDIM + k] * oW2[k];
  out[g] = acc;
  out[NG + g] = 1.f / (1.f + __expf(-acc));
}

// ---------------- driver ------------------------------------------------------
extern "C" void kernel_launch(void* const* d_in, const int* in_sizes, int n_in,
                              void* d_out, int out_size, void* d_ws, size_t ws_size,
                              hipStream_t stream) {
  (void)in_sizes; (void)n_in; (void)out_size; (void)ws_size;
  const float* x     = (const float*)d_in[0];
  const int*   ei    = (const int*)d_in[1];
  const int*   batch = (const int*)d_in[2];
  const float* eattr = (const float*)d_in[3];
  const float *nW1 = (const float*)d_in[4],  *nb1 = (const float*)d_in[5];
  const float *nW2 = (const float*)d_in[6],  *nb2 = (const float*)d_in[7];
  const float *nW3 = (const float*)d_in[8],  *nb3 = (const float*)d_in[9];
  const float *eW1 = (const float*)d_in[10], *eb1 = (const float*)d_in[11];
  const float *eW2 = (const float*)d_in[12], *eb2 = (const float*)d_in[13];
  const float *eW3 = (const float*)d_in[14], *eb3 = (const float*)d_in[15];
  const float* geps = (const float*)d_in[16];
  const float *gW1 = (const float*)d_in[17], *gb1 = (const float*)d_in[18];
  const float *gW2 = (const float*)d_in[19], *gb2 = (const float*)d_in[20];
  const float *oW1 = (const float*)d_in[21], *ob1 = (const float*)d_in[22];
  const float *ogm = (const float*)d_in[23], *obt = (const float*)d_in[24];
  const float *oW2 = (const float*)d_in[25], *ob2 = (const float*)d_in[26];
  const int* srcI = ei;
  const int* dstI = ei + N_EDGESC;

  char* base = (char*)d_ws;
  size_t off = 0;
  auto alloc = [&](size_t b) {
    char* p = base + off;
    off = (off + b + 255) & ~(size_t)255;
    return p;
  };
  float*    hbuf   = (float*)alloc((size_t)N_NODESC * HD * 4);     // 25.6 MB
  _Float16* eabuf  = (_Float16*)alloc((size_t)N_EDGESC * HD * 2);  // 204.8 MB
  _Float16* zbuf   = (_Float16*)alloc((size_t)N_NODESC * HD * 2);  // 12.8 MB
  int*      deg    = (int*)alloc((size_t)(N_NODESC + 1) * 4);
  int*      rowptr = (int*)alloc((size_t)(N_NODESC + 1) * 4);
  int*      tmpN   = (int*)alloc((size_t)N_NODESC * 4);            // excl / cursor
  int*      eidb   = (int*)alloc((size_t)N_EDGESC * 4);            // 6.4 MB
  int*      parts  = (int*)alloc(1024 * 4);
  unsigned* pk     = (unsigned*)alloc((size_t)NG * GDIM * 4);
  float*    ybuf   = (float*)alloc((size_t)NG * GDIM * 4);

  const int nodeTiles = N_NODESC / 16;                 // 6250
  const int edgeTiles = N_EDGESC / 16;                 // 100000
  const int nodeBlocks = min((nodeTiles + 7) / 8, GRID_BLK);
  const int edgeBlocks = min((edgeTiles + 7) / 8, GRID_BLK);
  const int scanBlocks = (N_NODESC + 1023) / 1024;     // 98

  // 1) embeddings (WMMA)
  embed_mlp_kernel<4, false><<<nodeBlocks, 256, 0, stream>>>(
      x, nW1, nb1, nW2, nb2, nW3, nb3, hbuf, (_Float16*)nullptr, N_NODESC);
  embed_mlp_kernel<8, true><<<edgeBlocks, 256, 0, stream>>>(
      eattr, eW1, eb1, eW2, eb2, eW3, eb3, (float*)nullptr, eabuf, N_EDGESC);

  // 2) CSR by destination (amortizes segment_max over the 3 layers)
  init_kernel<<<(N_NODESC + 256) / 256 + 1, 256, 0, stream>>>(deg, pk);
  count_kernel<<<(N_EDGESC + 255) / 256, 256, 0, stream>>>(dstI, deg);
  scan1_kernel<<<scanBlocks, 1024, 0, stream>>>(deg, tmpN, parts);
  scan2_kernel<<<1, 1024, 0, stream>>>(parts, scanBlocks);
  scan3_kernel<<<scanBlocks, 1024, 0, stream>>>(tmpN, parts, rowptr);
  cursor_kernel<<<(N_NODESC + 255) / 256, 256, 0, stream>>>(rowptr, tmpN);
  fill_kernel<<<(N_EDGESC + 255) / 256, 256, 0, stream>>>(dstI, tmpN, eidb);

  // 3) GINE layers
  for (int l = 0; l < 3; ++l) {
    agg_kernel<<<(N_NODESC + 7) / 8, 256, 0, stream>>>(rowptr, eidb, srcI, eabuf,
                                                       hbuf, geps, l, zbuf);
    gine_mlp_kernel<<<nodeBlocks, 256, 0, stream>>>(
        zbuf, gW1 + (size_t)l * HD * HD, gb1 + (size_t)l * HD,
        gW2 + (size_t)l * HD * HD, gb2 + (size_t)l * HD, hbuf, N_NODESC);
    pool_kernel<<<(int)(((size_t)N_NODESC * 32 + 255) / 256), 256, 0, stream>>>(
        hbuf, batch, pk, l);
  }

  // 4) output head
  out1_kernel<<<NG, GDIM, 0, stream>>>(pk, oW1, ob1, ogm, obt, ybuf);
  out2_kernel<<<1, NG, 0, stream>>>(ybuf, oW2, ob2, (float*)d_out);
}
